// GAT_47459388621528
// MI455X (gfx1250) — matrix operand order repaired and checked by reading, not compiled, not run on hardware
//
#include <hip/hip_runtime.h>
#include <hip/hip_bf16.h>

typedef __attribute__((ext_vector_type(16))) _Float16 v16h;
typedef __attribute__((ext_vector_type(8)))  _Float16 v8h;
typedef __attribute__((ext_vector_type(8)))  float    v8f;

#define TPB 256

// ---------------------------------------------------------------------------
// f32 [rows,D] -> f16 [rowsPad,D], zero-fill pad rows
// ---------------------------------------------------------------------------
__global__ void cvt_pad_f16(const float* __restrict__ src,
                            _Float16* __restrict__ dst,
                            int rows, int rowsPad, int D) {
    int i = blockIdx.x * blockDim.x + threadIdx.x;
    if (i >= rowsPad * D) return;
    int r = i / D;
    dst[i] = (r < rows) ? (_Float16)src[i] : (_Float16)0.f;
}

// ---------------------------------------------------------------------------
// Register-tiled WMMA GEMM: D[Mpad,colsPad] = A[Mpad,256] x B[colsPad,256]^T
// (+C) (optionally ELU).  One wave computes a 64x32 output block:
// 4 M-tiles x 2 N-tiles = 8 accumulators, 8 v_wmma per 32-wide K step,
// 12 b128 operand loads per step (1.5 loads/WMMA).
// __launch_bounds__(256,2): allow a large VGPR budget (64 acc VGPRs + staging)
// so the accumulators are never spilled to scratch.
// A layout per lane (16-bit A 16x32): elems 0..15 = d0 + 16*half + {0..15}
// B layout per lane (16-bit B 32x16): elems 0..15 = d0 + 16*half + {0..15}
// C/D layout: VGPR v -> row v + 8*half, col = lane & 15.
// ---------------------------------------------------------------------------
__global__ void __launch_bounds__(256, 2)
gat_wmma_gemm(const _Float16* __restrict__ A,
              const _Float16* __restrict__ B,
              const float* __restrict__ C,
              float* __restrict__ D,
              int Mtiles4, int colsPad, int addc, int do_elu) {
    const int lane   = threadIdx.x & 31;
    const int wave   = blockIdx.x * (blockDim.x >> 5) + (threadIdx.x >> 5);
    const int tilesN2 = colsPad >> 5;               // 32-col groups
    if (wave >= Mtiles4 * tilesN2) return;          // wave-uniform exit
    const int tm   = wave / tilesN2;                // 64-row group
    const int tn   = wave - tm * tilesN2;           // 32-col group
    const int r    = lane & 15;
    const int half = lane >> 4;

    const _Float16* ar_[4];
#pragma unroll
    for (int i = 0; i < 4; ++i)
        ar_[i] = A + (size_t)(tm * 64 + i * 16 + r) * 256;
    const _Float16* br_[2];
#pragma unroll
    for (int j = 0; j < 2; ++j)
        br_[j] = B + (size_t)(tn * 32 + j * 16 + r) * 256;

    v8f acc[4][2];
#pragma unroll
    for (int i = 0; i < 4; ++i)
#pragma unroll
        for (int j = 0; j < 2; ++j) {
            if (addc) {
#pragma unroll
                for (int v = 0; v < 8; ++v)
                    acc[i][j][v] = C[(size_t)(tm * 64 + i * 16 + v + 8 * half) * colsPad
                                     + tn * 32 + j * 16 + r];
            } else {
                acc[i][j] = (v8f){0.f, 0.f, 0.f, 0.f, 0.f, 0.f, 0.f, 0.f};
            }
        }

#pragma unroll
    for (int d0 = 0; d0 < 256; d0 += 32) {
        // load both B tiles for this K step (4 x b128)
        v16h bv[2];
#pragma unroll
        for (int j = 0; j < 2; ++j) {
            v8h lo = *(const v8h*)(br_[j] + d0 + 16 * half);
            v8h hi = *(const v8h*)(br_[j] + d0 + 16 * half + 8);
            bv[j] = __builtin_shufflevector(lo, hi, 0,1,2,3,4,5,6,7,8,9,10,11,12,13,14,15);
        }
        // per A tile: load (2 x b128) then 2 WMMAs -> short A live ranges
#pragma unroll
        for (int i = 0; i < 4; ++i) {
            v8h lo = *(const v8h*)(ar_[i] + d0 + 16 * half);
            v8h hi = *(const v8h*)(ar_[i] + d0 + 16 * half + 8);
            v16h av = __builtin_shufflevector(lo, hi, 0,1,2,3,4,5,6,7,8,9,10,11,12,13,14,15);
#pragma unroll
            for (int j = 0; j < 2; ++j)
                acc[i][j] = __builtin_amdgcn_wmma_f32_16x16x32_f16(
                    false, av, false, bv[j], (short)0, acc[i][j], false, false);
        }
    }

#pragma unroll
    for (int i = 0; i < 4; ++i)
#pragma unroll
        for (int j = 0; j < 2; ++j)
#pragma unroll
            for (int v = 0; v < 8; ++v) {
                float val = acc[i][j][v];
                if (do_elu) val = val > 0.f ? val : (__expf(val) - 1.f);
                D[(size_t)(tm * 64 + i * 16 + v + 8 * half) * colsPad
                  + tn * 32 + j * 16 + r] = val;
            }
}

// ---------------------------------------------------------------------------
// a1[n,h] = sum_k ft[n,h*K+k]*al[h,k]; a2 likewise; init smax=-inf, denom=0
// ---------------------------------------------------------------------------
__global__ void attn_prep(const float* __restrict__ ft,
                          const float* __restrict__ al,
                          const float* __restrict__ ar,
                          float* __restrict__ a1, float* __restrict__ a2,
                          float* __restrict__ smax, float* __restrict__ denom,
                          int N, int H, int K, int stride) {
    int i = blockIdx.x * blockDim.x + threadIdx.x;
    if (i >= N * H) return;
    int n = i / H, h = i - n * H;
    const float* f = ft + (size_t)n * stride + h * K;
    const float* L = al + h * K;
    const float* R = ar + h * K;
    float s1 = 0.f, s2 = 0.f;
    for (int k = 0; k < K; ++k) {
        float v = f[k];
        s1 += v * L[k];
        s2 += v * R[k];
    }
    a1[i] = s1;
    a2[i] = s2;
    smax[i]  = __uint_as_float(0xFF800000u);   // -inf
    denom[i] = 0.f;
}

// order-preserving float atomic max via signed/unsigned int atomics
__device__ inline void atomicMaxF(float* addr, float v) {
    if (v >= 0.f) atomicMax((int*)addr, __float_as_int(v));
    else          atomicMin((unsigned int*)addr, __float_as_uint(v));
}

// per-edge score: s = leaky_relu(a1[dst]+a2[src]); segment max over dst
__global__ void edge_score(const int* __restrict__ src, const int* __restrict__ dst,
                           const float* __restrict__ a1, const float* __restrict__ a2,
                           float* __restrict__ s, float* __restrict__ smax,
                           int E, int H) {
    int i = blockIdx.x * blockDim.x + threadIdx.x;
    if (i >= E * H) return;
    int e = i / H, h = i - e * H;
    float v = a1[dst[e] * H + h] + a2[src[e] * H + h];
    v = v > 0.f ? v : 0.01f * v;
    s[i] = v;
    atomicMaxF(&smax[dst[e] * H + h], v);
}

// ex = exp(s - smax[dst]); denom[dst] += ex   (s overwritten with ex)
__global__ void edge_exp(const int* __restrict__ dst, float* __restrict__ s,
                         const float* __restrict__ smax, float* __restrict__ denom,
                         int E, int H) {
    int i = blockIdx.x * blockDim.x + threadIdx.x;
    if (i >= E * H) return;
    int e = i / H, h = i - e * H;
    float ex = __expf(s[i] - smax[dst[e] * H + h]);
    s[i] = ex;
    atomicAdd(&denom[dst[e] * H + h], ex);
}

// accum[dst, h*K+k] += ex * ft[src, h*K+k]
__global__ void edge_msg(const int* __restrict__ src, const int* __restrict__ dst,
                         const float* __restrict__ s, const float* __restrict__ ft,
                         float* __restrict__ accum, int E, int H, int K, int stride) {
    int i = blockIdx.x * blockDim.x + threadIdx.x;
    if (i >= E * H) return;
    int e = i / H, h = i - e * H;
    float ex = s[i];
    const float* f = ft + (size_t)src[e] * stride + h * K;
    float* a       = accum + (size_t)dst[e] * stride + h * K;
    for (int k = 0; k < K; ++k) atomicAdd(a + k, ex * f[k]);
}

// out[n,c] = accum[n,c]/denom (1 if denom==0)  (optional ELU)
__global__ void div_norm(const float* __restrict__ accum, const float* __restrict__ denom,
                         float* __restrict__ out, int N, int cols, int stride,
                         int outstride, int H, int K, int do_elu) {
    int i = blockIdx.x * blockDim.x + threadIdx.x;
    if (i >= N * cols) return;
    int n = i / cols, c = i - n * cols;
    int h = c / K;
    float d = denom[n * H + h];
    d = d > 0.f ? d : 1.f;
    float val = accum[(size_t)n * stride + c] / d;
    if (do_elu) val = val > 0.f ? val : (__expf(val) - 1.f);
    out[(size_t)n * outstride + c] = val;
}

// out[n,c] = mean_h hf[n, h*121+c]   (hf stride 736)
__global__ void avg_heads(const float* __restrict__ hf, float* __restrict__ out, int N) {
    int i = blockIdx.x * blockDim.x + threadIdx.x;
    if (i >= N * 121) return;
    int n = i / 121, c = i - n * 121;
    float sum = 0.f;
#pragma unroll
    for (int h = 0; h < 6; ++h) sum += hf[(size_t)n * 736 + h * 121 + c];
    out[i] = sum * (1.f / 6.f);
}

// ---------------------------------------------------------------------------
extern "C" void kernel_launch(void* const* d_in, const int* in_sizes, int n_in,
                              void* d_out, int out_size, void* d_ws, size_t ws_size,
                              hipStream_t stream) {
    const float* features = (const float*)d_in[0];
    const int*   srcI     = (const int*)d_in[1];
    const int*   dstI     = (const int*)d_in[2];
    const float* W0  = (const float*)d_in[3];
    const float* al0 = (const float*)d_in[4];
    const float* ar0 = (const float*)d_in[5];
    const float* W1  = (const float*)d_in[6];
    const float* al1 = (const float*)d_in[7];
    const float* ar1 = (const float*)d_in[8];
    const float* R1  = (const float*)d_in[9];
    const float* Wf  = (const float*)d_in[10];
    const float* alf = (const float*)d_in[11];
    const float* arf = (const float*)d_in[12];
    const float* Rf  = (const float*)d_in[13];
    float* out = (float*)d_out;

    const int D = 256;
    const int N = in_sizes[0] / D;                 // 50000
    const int E = in_sizes[1];                     // 400000
    const int Mpad = ((N + 63) / 64) * 64;         // 50048 (64-row GEMM blocks)
    const int CF = 726, CFP = 736;                 // final cols / padded

    // ---- carve workspace ----
    char* ws = (char*)d_ws;
    size_t off = 0;
    auto carve = [&](size_t bytes) -> void* {
        void* p = ws + off;
        off += (bytes + 255) & ~(size_t)255;
        return p;
    };
    _Float16* x16 = (_Float16*)carve((size_t)Mpad * D * 2);
    _Float16* w16 = (_Float16*)carve((size_t)CFP * D * 2);
    _Float16* r16 = (_Float16*)carve((size_t)CFP * D * 2);
    float* ft    = (float*)carve((size_t)Mpad * CFP * 4);
    float* accum = (float*)carve((size_t)Mpad * CFP * 4);
    float* a1    = (float*)carve((size_t)N * 6 * 4);
    float* a2    = (float*)carve((size_t)N * 6 * 4);
    float* smax  = (float*)carve((size_t)N * 6 * 4);
    float* denom = (float*)carve((size_t)N * 6 * 4);
    float* sbuf  = (float*)carve((size_t)E * 6 * 4);
    float* xa    = (float*)carve((size_t)Mpad * D * 4);

    auto cdiv = [](long a, long b) { return (int)((a + b - 1) / b); };

    auto gemm = [&](const _Float16* A, const _Float16* B, const float* C,
                    float* Dp, int colsPad, int addc, int elu) {
        long waves = (long)(Mpad / 64) * (colsPad / 32);
        gat_wmma_gemm<<<cdiv(waves, 8), 256, 0, stream>>>(A, B, C, Dp, Mpad / 64,
                                                          colsPad, addc, elu);
    };
    auto run_attention = [&](const float* al, const float* ar, int H, int K,
                             int stride) {
        hipMemsetAsync(accum, 0, (size_t)Mpad * stride * 4, stream);
        attn_prep<<<cdiv((long)N * H, TPB), TPB, 0, stream>>>(ft, al, ar, a1, a2,
                                                              smax, denom, N, H, K, stride);
        edge_score<<<cdiv((long)E * H, TPB), TPB, 0, stream>>>(srcI, dstI, a1, a2,
                                                               sbuf, smax, E, H);
        edge_exp<<<cdiv((long)E * H, TPB), TPB, 0, stream>>>(dstI, sbuf, smax, denom, E, H);
        edge_msg<<<cdiv((long)E * H, TPB), TPB, 0, stream>>>(srcI, dstI, sbuf, ft,
                                                             accum, E, H, K, stride);
    };

    // ================= Layer 0: 4 heads, 256 -> 4x64, no residual ============
    cvt_pad_f16<<<cdiv((long)Mpad * D, TPB), TPB, 0, stream>>>(features, x16, N, Mpad, D);
    cvt_pad_f16<<<cdiv(256L * D, TPB), TPB, 0, stream>>>(W0, w16, 256, 256, D);
    gemm(x16, w16, nullptr, ft, 256, 0, 0);
    run_attention(al0, ar0, 4, 64, 256);
    div_norm<<<cdiv((long)N * 256, TPB), TPB, 0, stream>>>(accum, denom, xa, N, 256,
                                                           256, 256, 4, 64, 1);

    // ================= Layer 1: 4 heads, 256 -> 4x64, residual via R1 ========
    cvt_pad_f16<<<cdiv((long)Mpad * D, TPB), TPB, 0, stream>>>(xa, x16, N, Mpad, D);
    cvt_pad_f16<<<cdiv(256L * D, TPB), TPB, 0, stream>>>(W1, w16, 256, 256, D);
    gemm(x16, w16, nullptr, ft, 256, 0, 0);
    run_attention(al1, ar1, 4, 64, 256);
    div_norm<<<cdiv((long)N * 256, TPB), TPB, 0, stream>>>(accum, denom, accum, N, 256,
                                                           256, 256, 4, 64, 0);
    cvt_pad_f16<<<cdiv(256L * D, TPB), TPB, 0, stream>>>(R1, r16, 256, 256, D);
    gemm(x16, r16, accum, xa, 256, 1, 1);   // xa = elu(x@R1^T + accum)

    // ================= Final: 6 heads, 256 -> 6x121, residual via Rf =========
    cvt_pad_f16<<<cdiv((long)Mpad * D, TPB), TPB, 0, stream>>>(xa, x16, N, Mpad, D);
    cvt_pad_f16<<<cdiv((long)CFP * D, TPB), TPB, 0, stream>>>(Wf, w16, CF, CFP, D);
    gemm(x16, w16, nullptr, ft, CFP, 0, 0);
    run_attention(alf, arf, 6, 121, CFP);
    div_norm<<<cdiv((long)N * CF, TPB), TPB, 0, stream>>>(accum, denom, accum, N, CF,
                                                          CFP, CFP, 6, 121, 0);
    cvt_pad_f16<<<cdiv((long)CFP * D, TPB), TPB, 0, stream>>>(Rf, r16, CF, CFP, D);
    gemm(x16, r16, accum, ft, CFP, 1, 1);   // ft = hf = elu(x@Rf^T + accum)
    avg_heads<<<cdiv((long)N * 121, TPB), TPB, 0, stream>>>(ft, out, N);
}
